// YoloV3_26603027432031
// MI455X (gfx1250) — compile-verified, matrix operand order checked by reading, not run on hardware
//
#include <hip/hip_runtime.h>
#include <math.h>

#define BATCH 16
#define NCLS  80
#define N1    1083      // 19*19*3
#define N2    4332      // 38*38*3
#define N3    17328     // 76*76*3
#define NBOX  22743     // N1+N2+N3
#define KPC   256
#define NROWS (BATCH*NCLS)   // 1280
#define FLAT  (NCLS*KPC)     // 20480
#define MAXT  100

typedef float v2f __attribute__((ext_vector_type(2)));
typedef float v8f __attribute__((ext_vector_type(8)));
typedef _Float16 v16h __attribute__((ext_vector_type(16)));

__device__ __constant__ float ANC_W[9] = {
  10.f/416.f, 16.f/416.f, 33.f/416.f, 30.f/416.f, 62.f/416.f,
  59.f/416.f, 116.f/416.f, 156.f/416.f, 373.f/416.f };
__device__ __constant__ float ANC_H[9] = {
  13.f/416.f, 30.f/416.f, 23.f/416.f, 61.f/416.f, 45.f/416.f,
  119.f/416.f, 90.f/416.f, 198.f/416.f, 326.f/416.f };

__device__ __forceinline__ float sigmoidf(float x) {
  return 1.f / (1.f + expf(-x));
}

// ---------------------------------------------------------------- kernel 1
// decode: boxes[b][n][4] (x1,y1,x2,y2) and objectness[b][n]
__global__ void __launch_bounds__(256)
decode_kernel(const float* __restrict__ p1, const float* __restrict__ p2,
              const float* __restrict__ p3, float* __restrict__ boxes,
              float* __restrict__ obj) {
  int t = blockIdx.x * blockDim.x + threadIdx.x;
  if (t >= BATCH * NBOX) return;
  int b = t / NBOX, n = t % NBOX;
  const float* q; int G, m, abase;
  if (n < N1)          { q = p1 + ((size_t)b * N1 + n) * 85;            G = 19; m = n;          abase = 6; }
  else if (n < N1+N2)  { q = p2 + ((size_t)b * N2 + (n - N1)) * 85;     G = 38; m = n - N1;     abase = 3; }
  else                 { q = p3 + ((size_t)b * N3 + (n - N1 - N2)) * 85;G = 76; m = n - N1 - N2;abase = 0; }
  int a = m % 3, cell = m / 3;
  int gx = cell % G, gy = cell / G;
  float invG = 1.f / (float)G;
  float x = (sigmoidf(q[0]) + (float)gx) * invG;
  float y = (sigmoidf(q[1]) + (float)gy) * invG;
  float w = expf(q[2]) * ANC_W[abase + a];
  float h = expf(q[3]) * ANC_H[abase + a];
  float4 bb;
  bb.x = x - 0.5f * w; bb.y = y - 0.5f * h;
  bb.z = x + 0.5f * w; bb.w = y + 0.5f * h;
  ((float4*)boxes)[t] = bb;
  obj[t] = sigmoidf(q[4]);
}

// ---------------------------------------------------------------- kernel 2
// per-(batch,class) top-256 with index tie-break (lowest index first).
// Scores recomputed on the fly (obj * sigmoid(cls logit)) -> no 116MB tensor.
__global__ void __launch_bounds__(256)
topk_kernel(const float* __restrict__ p1, const float* __restrict__ p2,
            const float* __restrict__ p3, const float* __restrict__ obj,
            const float* __restrict__ boxes, float* __restrict__ selScores,
            int* __restrict__ selIdx, float* __restrict__ selBoxes) {
  extern __shared__ float buf[];                 // NBOX floats (~89KB of 320KB LDS)
  __shared__ float rS[8];  __shared__ int rI[8];
  __shared__ float selS_s[KPC]; __shared__ int selI_s[KPC];
  __shared__ int winI;

  int bc = blockIdx.x;
  int b = bc / NCLS, c = bc % NCLS;
  int tid = threadIdx.x, lane = tid & 31, wv = tid >> 5;

  for (int n = tid; n < NBOX; n += 256) {
    const float* q;
    if (n < N1)         q = p1 + ((size_t)b * N1 + n) * 85;
    else if (n < N1+N2) q = p2 + ((size_t)b * N2 + (n - N1)) * 85;
    else                q = p3 + ((size_t)b * N3 + (n - N1 - N2)) * 85;
    buf[n] = obj[b * NBOX + n] * sigmoidf(q[5 + c]);
  }
  __syncthreads();

  // incremental local maxima: only the winner's owner rescans its slots
  float sLoc = -__builtin_inff(); int iLoc = 0x7FFFFFFF;
  for (int n = tid; n < NBOX; n += 256) {
    float v = buf[n];
    if (v > sLoc) { sLoc = v; iLoc = n; }
  }
  for (int k = 0; k < KPC; ++k) {
    float s = sLoc; int i = iLoc;
    #pragma unroll
    for (int off = 16; off > 0; off >>= 1) {
      float s2 = __shfl_down(s, off, 32);
      int   i2 = __shfl_down(i, off, 32);
      if (s2 > s || (s2 == s && i2 < i)) { s = s2; i = i2; }
    }
    if (lane == 0) { rS[wv] = s; rI[wv] = i; }
    __syncthreads();
    if (wv == 0) {
      float s2 = (lane < 8) ? rS[lane] : -__builtin_inff();
      int   i2 = (lane < 8) ? rI[lane] : 0x7FFFFFFF;
      #pragma unroll
      for (int off = 4; off > 0; off >>= 1) {
        float s3 = __shfl_down(s2, off, 32);
        int   i3 = __shfl_down(i2, off, 32);
        if (s3 > s2 || (s3 == s2 && i3 < i2)) { s2 = s3; i2 = i3; }
      }
      if (lane == 0) {
        selS_s[k] = s2; selI_s[k] = i2;
        winI = i2;
        buf[i2] = -__builtin_inff();
      }
    }
    __syncthreads();
    if (iLoc == winI) {                 // exactly one owner: rescan 89 slots
      sLoc = -__builtin_inff(); iLoc = 0x7FFFFFFF;
      for (int n = tid; n < NBOX; n += 256) {
        float v = buf[n];
        if (v > sLoc) { sLoc = v; iLoc = n; }
      }
    }
  }
  // write results + gather boxes
  int idx = selI_s[tid];
  selScores[bc * KPC + tid] = selS_s[tid];
  selIdx[bc * KPC + tid]    = idx;
  ((float4*)selBoxes)[bc * KPC + tid] = ((const float4*)boxes)[b * NBOX + idx];
}

// ---------------------------------------------------------------- kernel 3
// NMS per (batch,class). 16x16 tiles: area-sum a_i + a_j via one
// V_WMMA_F32_16X16X4_F32 (A=[a_i,1,0,0], B=[1;a_j;0;0]); intersection via
// VALU min/max. "iou > 0.5" is evaluated division-free as 3*inter > a_i+a_j
// (valid since union = sum-inter >= max(a_i,a_j) > 0). Ballot-packed into a
// 256x256 bitmask; greedy pass runs barrier-free on one wave (lanes 0..7
// carry the 256-bit removed mask); per-class cap via wave32 ballot prefix.
__global__ void __launch_bounds__(256)
nms_kernel(const float* __restrict__ selScores, const float* __restrict__ selBoxes,
           float* __restrict__ finalScores) {
  __shared__ float4 bx[KPC];
  __shared__ float  area[KPC];
  __shared__ unsigned int sup[KPC * 8];   // 256 rows x 8 words
  __shared__ int kept[KPC];
  __shared__ unsigned int remWords[8];
  __shared__ int wsum[8];

  int bc = blockIdx.x, tid = threadIdx.x;
  int lane = tid & 31, wv = tid >> 5;

  float4 bb = ((const float4*)selBoxes)[bc * KPC + tid];
  bx[tid] = bb;
  area[tid] = (bb.z - bb.x) * (bb.w - bb.y);
  float myScore = selScores[bc * KPC + tid];
  kept[tid] = (myScore > 0.5f) ? 1 : 0;
  #pragma unroll
  for (int w = tid; w < KPC * 8; w += 256) sup[w] = 0u;
#if __has_builtin(__builtin_amdgcn_s_wait_tensorcnt)
  __builtin_amdgcn_s_wait_tensorcnt(0);   // CDNA5 split-counter path
#endif
  __syncthreads();

  for (int t = wv; t < 256; t += 8) {     // 32 tiles per wave, uniform per wave
    int ti = t >> 4, tj = t & 15;
    v8f d;
#if __has_builtin(__builtin_amdgcn_wmma_f32_16x16x4_f32)
    v2f av = { (lane < 16) ? area[ti * 16 + lane] : 0.f,
               (lane < 16) ? 1.f : 0.f };
    v2f bv = { (lane < 16) ? 1.f : 0.f,
               (lane < 16) ? area[tj * 16 + lane] : 0.f };
    v8f cz = {0.f,0.f,0.f,0.f,0.f,0.f,0.f,0.f};
    d = __builtin_amdgcn_wmma_f32_16x16x4_f32(false, av, false, bv,
                                              (short)0, cz, false, false);
#else
    v16h ah, bh;
    #pragma unroll
    for (int z = 0; z < 16; ++z) { ah[z] = (_Float16)0.f; bh[z] = (_Float16)0.f; }
    ah[0] = (lane < 16) ? (_Float16)area[ti * 16 + lane] : (_Float16)0.f;
    ah[1] = (lane < 16) ? (_Float16)1.f : (_Float16)0.f;
    bh[0] = (lane < 16) ? (_Float16)1.f : (_Float16)0.f;
    bh[1] = (lane < 16) ? (_Float16)area[tj * 16 + lane] : (_Float16)0.f;
    v8f cz = {0.f,0.f,0.f,0.f,0.f,0.f,0.f,0.f};
    d = __builtin_amdgcn_wmma_f32_16x16x32_f16(false, ah, false, bh,
                                               (short)0, cz, false, false);
#endif
    int n = tj * 16 + (lane & 15);                   // col (fixed per lane)
    float4 bj = bx[n];
    #pragma unroll
    for (int r = 0; r < 8; ++r) {
      int m = ti * 16 + ((lane < 16) ? r : r + 8);   // row (C/D VGPR layout)
      float4 bi = bx[m];
      float iw = fmaxf(fminf(bi.z, bj.z) - fmaxf(bi.x, bj.x), 0.f);
      float ih = fmaxf(fminf(bi.w, bj.w) - fmaxf(bi.y, bj.y), 0.f);
      float inter = iw * ih;
      // iou > 0.5  <=>  inter > 0.5*(sum-inter)  <=>  3*inter > sum
      bool pred = (3.f * inter > d[r]);
#if __has_builtin(__builtin_amdgcn_ballot_w32)
      unsigned int bal = __builtin_amdgcn_ballot_w32(pred);
#else
      unsigned int bal = (unsigned int)__ballot(pred);
#endif
      if (lane == 0) {
        unsigned int lo = bal & 0xFFFFu, hi = bal >> 16;
        int shift = (tj & 1) * 16;
        atomicOr(&sup[(ti * 16 + r)     * 8 + (tj >> 1)], lo << shift);
        atomicOr(&sup[(ti * 16 + 8 + r) * 8 + (tj >> 1)], hi << shift);
      }
    }
  }
  __syncthreads();

  // barrier-free greedy suppression on wave 0: lanes 0..7 hold the 256-bit
  // "removed" mask; only bits j>i of row i are applied (diag/symmetric bits
  // must not retro-suppress earlier boxes).
  if (wv == 0) {
    unsigned int rem = 0u;
    for (int i = 0; i < KPC; ++i) {
      unsigned int wrd = (unsigned int)__shfl((int)rem, i >> 5, 32);
      bool alive = kept[i] && !((wrd >> (i & 31)) & 1u);
      if (alive) {
        unsigned int sw = (lane < 8) ? sup[i * 8 + lane] : 0u;
        int iw_ = i >> 5, ib = i & 31;
        unsigned int msk;
        if (lane < iw_)       msk = 0u;
        else if (lane == iw_) msk = (ib == 31) ? 0u : (0xFFFFFFFFu << (ib + 1));
        else                  msk = 0xFFFFFFFFu;
        rem |= sw & msk;
      }
    }
    if (lane < 8) remWords[lane] = rem;
  }
  __syncthreads();

  int keepFlag = kept[tid] &&
                 !((remWords[tid >> 5] >> (tid & 31)) & 1u);

  // inclusive prefix count via wave32 ballot; cap at MAX_PER_CLASS=100
#if __has_builtin(__builtin_amdgcn_ballot_w32)
  unsigned int kbal = __builtin_amdgcn_ballot_w32(keepFlag != 0);
#else
  unsigned int kbal = (unsigned int)__ballot(keepFlag != 0);
#endif
  int preInWave = __popc(kbal & ((lane == 0) ? 0u : (0xFFFFFFFFu >> (32 - lane))));
  if (lane == 0) wsum[wv] = __popc(kbal);
  __syncthreads();
  int waveBase = 0;
  #pragma unroll
  for (int w = 0; w < 8; ++w) waveBase += (w < wv) ? wsum[w] : 0;
  int incl = waveBase + preInWave + keepFlag;
  int keep = keepFlag && (incl <= 100);
  finalScores[bc * KPC + tid] = keep ? myScore : -1.0f;
}

// ---------------------------------------------------------------- kernel 4
// final top-100 per batch over 20480 LDS-resident scores; write outputs.
__global__ void __launch_bounds__(256)
final_topk_kernel(const float* __restrict__ finalScores,
                  const float* __restrict__ selBoxes, float* __restrict__ out) {
  extern __shared__ float buf[];                 // FLAT floats (80KB)
  __shared__ float rS[8]; __shared__ int rI[8];
  __shared__ float selS[MAXT]; __shared__ int selI[MAXT];
  __shared__ int winI;
  __shared__ int cnt;

  int b = blockIdx.x, tid = threadIdx.x;
  int lane = tid & 31, wv = tid >> 5;

  for (int n = tid; n < FLAT; n += 256) buf[n] = finalScores[b * FLAT + n];
  if (tid == 0) cnt = 0;
  __syncthreads();

  float sLoc = -__builtin_inff(); int iLoc = 0x7FFFFFFF;
  for (int n = tid; n < FLAT; n += 256) {
    float v = buf[n];
    if (v > sLoc) { sLoc = v; iLoc = n; }
  }
  for (int k = 0; k < MAXT; ++k) {
    float s = sLoc; int i = iLoc;
    #pragma unroll
    for (int off = 16; off > 0; off >>= 1) {
      float s2 = __shfl_down(s, off, 32);
      int   i2 = __shfl_down(i, off, 32);
      if (s2 > s || (s2 == s && i2 < i)) { s = s2; i = i2; }
    }
    if (lane == 0) { rS[wv] = s; rI[wv] = i; }
    __syncthreads();
    if (wv == 0) {
      float s2 = (lane < 8) ? rS[lane] : -__builtin_inff();
      int   i2 = (lane < 8) ? rI[lane] : 0x7FFFFFFF;
      #pragma unroll
      for (int off = 4; off > 0; off >>= 1) {
        float s3 = __shfl_down(s2, off, 32);
        int   i3 = __shfl_down(i2, off, 32);
        if (s3 > s2 || (s3 == s2 && i3 < i2)) { s2 = s3; i2 = i3; }
      }
      if (lane == 0) {
        selS[k] = s2; selI[k] = i2; winI = i2;
        buf[i2] = -__builtin_inff();
      }
    }
    __syncthreads();
    if (iLoc == winI) {
      sLoc = -__builtin_inff(); iLoc = 0x7FFFFFFF;
      for (int n = tid; n < FLAT; n += 256) {
        float v = buf[n];
        if (v > sLoc) { sLoc = v; iLoc = n; }
      }
    }
  }
  if (tid < MAXT) {
    float s = selS[tid]; int idx = selI[tid];
    int valid = (s > 0.f) ? 1 : 0;
    float4 bb = ((const float4*)selBoxes)[(size_t)b * FLAT + idx];
    float* ob = out + ((size_t)b * MAXT + tid) * 4;
    ob[0] = valid ? bb.x : 0.f; ob[1] = valid ? bb.y : 0.f;
    ob[2] = valid ? bb.z : 0.f; ob[3] = valid ? bb.w : 0.f;
    out[BATCH*MAXT*4 + b * MAXT + tid] = valid ? s : 0.f;                 // scores
    out[BATCH*MAXT*4 + BATCH*MAXT + b * MAXT + tid] =
        valid ? (float)(idx >> 8) : 0.f;                                  // classes
    if (valid) atomicAdd(&cnt, 1);
  }
  __syncthreads();
  if (tid == 0)
    ((int*)out)[BATCH*MAXT*4 + 2*BATCH*MAXT + b] = cnt;                   // valid
}

// ---------------------------------------------------------------- launch
extern "C" void kernel_launch(void* const* d_in, const int* in_sizes, int n_in,
                              void* d_out, int out_size, void* d_ws, size_t ws_size,
                              hipStream_t stream) {
  const float* p1 = (const float*)d_in[0];   // (16,19,19,3,85)
  const float* p2 = (const float*)d_in[1];   // (16,38,38,3,85)
  const float* p3 = (const float*)d_in[2];   // (16,76,76,3,85)

  float* ws          = (float*)d_ws;
  float* boxes       = ws;                                        // B*NBOX*4
  float* obj         = boxes + (size_t)BATCH * NBOX * 4;          // B*NBOX
  float* selScores   = obj + (size_t)BATCH * NBOX;                // 1280*256
  int*   selIdx      = (int*)(selScores + (size_t)NROWS * KPC);   // 1280*256
  float* selBoxes    = (float*)(selIdx + (size_t)NROWS * KPC);    // 1280*256*4
  float* finalScores = selBoxes + (size_t)NROWS * KPC * 4;        // 16*20480
  float* out         = (float*)d_out;

  int total = BATCH * NBOX;
  decode_kernel<<<(total + 255) / 256, 256, 0, stream>>>(p1, p2, p3, boxes, obj);
  topk_kernel<<<NROWS, 256, NBOX * sizeof(float), stream>>>(
      p1, p2, p3, obj, boxes, selScores, selIdx, selBoxes);
  nms_kernel<<<NROWS, 256, 0, stream>>>(selScores, selBoxes, finalScores);
  final_topk_kernel<<<BATCH, 256, FLAT * sizeof(float), stream>>>(
      finalScores, selBoxes, out);
}
